// FastHyperbolicAttention_18150531793021
// MI455X (gfx1250) — matrix operand order repaired and checked by reading, not compiled, not run on hardware
//
#include <hip/hip_runtime.h>

typedef __attribute__((ext_vector_type(16))) _Float16 v16h;
typedef __attribute__((ext_vector_type(8)))  _Float16 v8h;
typedef __attribute__((ext_vector_type(8)))  float    v8f;

#define HA_B  2
#define HA_N  2048
#define HA_D  512
#define HA_H  8
#define HA_DH 64
#define HA_BN (HA_B * HA_N)   // 4096

// ---- CDNA5 async global->LDS path (probe via __has_builtin; sync fallback) ----
#if defined(__has_builtin)
#  if __has_builtin(__builtin_amdgcn_global_load_async_to_lds_b128)
#    define HA_ASYNC 1
#  endif
#endif
#ifndef HA_ASYNC
#  define HA_ASYNC 0
#endif

#if HA_ASYNC
#  if defined(__has_builtin) && __has_builtin(__builtin_amdgcn_s_wait_asynccnt)
#    define HA_WAIT_ASYNC() __builtin_amdgcn_s_wait_asynccnt(0)
#  else
#    define HA_WAIT_ASYNC() asm volatile("s_wait_asynccnt 0" ::: "memory")
#  endif
// builtin signature (from toolchain diagnostic): (v4i as1*, v4i as3*, imm offset, imm cpol)
typedef int ha_v4i __attribute__((vector_size(16)));
typedef __attribute__((address_space(1))) ha_v4i ha_gv4i;
typedef __attribute__((address_space(3))) ha_v4i ha_lv4i;
// copy 16 halves (32 B) global -> LDS as two async b128 DMA transfers
__device__ __forceinline__ void ha_async_cp16h(const _Float16* g, _Float16* l) {
  __builtin_amdgcn_global_load_async_to_lds_b128((ha_gv4i*)g, (ha_lv4i*)l, 0, 0);
  __builtin_amdgcn_global_load_async_to_lds_b128((ha_gv4i*)g, (ha_lv4i*)l, 16, 0);
}
#endif

__device__ __forceinline__ v8f wmma_f16(v16h a, v16h b, v8f c) {
  return __builtin_amdgcn_wmma_f32_16x16x32_f16(false, a, false, b, (short)0, c, false, false);
}

__device__ __forceinline__ v16h cat8(v8h lo, v8h hi) {
  return __builtin_shufflevector(lo, hi, 0,1,2,3,4,5,6,7,8,9,10,11,12,13,14,15);
}

// ---------------------------------------------------------------- cast f32->f16
__global__ void hyp_cast(const float* __restrict__ s, _Float16* __restrict__ d, int n) {
  int i = blockIdx.x * blockDim.x + threadIdx.x;
  int stride = gridDim.x * blockDim.x;
  for (; i < n; i += stride) d[i] = (_Float16)s[i];
}

// ---------------------------------------------------------------- row norms
__global__ void hyp_norms(const _Float16* __restrict__ qh, const _Float16* __restrict__ kh,
                          float* __restrict__ qn, float* __restrict__ kn) {
  int i = blockIdx.x * blockDim.x + threadIdx.x;
  if (i >= HA_B * HA_H * HA_N) return;
  float s = 0.f, t = 0.f;
  for (int d = 0; d < HA_DH; d++) {
    float a = (float)qh[(size_t)i * HA_DH + d]; s += a * a;
    float b = (float)kh[(size_t)i * HA_DH + d]; t += b * b;
  }
  qn[i] = s; kn[i] = t;
}

// ---------------------------------------------------------------- WMMA GEMM
// out[m,n] = sum_k A[m,k] * W[n,k]
// mode 0: f16 [b,h,n,dh] | mode 1: f16 [b,h,dh,n] | mode 2: f32 [m,n]
#define GT_M 128
#define GT_N 128
#define GT_K 32
#define GT_NT (HA_D / GT_K)   // 16 K-tiles
#define LDT  40               // padded LDS stride in halves (80 B)

__device__ __forceinline__ void gemm_step(const _Float16* __restrict__ Ac,
                                          const _Float16* __restrict__ Bc,
                                          v8f acc[2][4],
                                          int wrow, int wcol, int ln, int half, int kb) {
  v16h af[2];
  for (int rt = 0; rt < 2; rt++) {
    const _Float16* p = &Ac[(wrow + rt * 16 + ln) * LDT + kb];
    af[rt] = cat8(*(const v8h*)p, *(const v8h*)(p + 16));
  }
  v16h bf[4];
  for (int ct = 0; ct < 4; ct++) {
    const _Float16* p = &Bc[(wcol + ct * 16 + ln) * LDT + half * 8];
    bf[ct] = cat8(*(const v8h*)p, *(const v8h*)(p + 16));
  }
  for (int rt = 0; rt < 2; rt++)
    for (int ct = 0; ct < 4; ct++)
      acc[rt][ct] = wmma_f16(af[rt], bf[ct], acc[rt][ct]);
}

__global__ __launch_bounds__(256)
void hyp_gemm(const _Float16* __restrict__ A, const _Float16* __restrict__ W,
              void* __restrict__ dst, int mode) {
#if HA_ASYNC
  __shared__ _Float16 As[2 * GT_M * LDT];
  __shared__ _Float16 Bs[2 * GT_N * LDT];
#else
  __shared__ _Float16 As[GT_M * LDT];
  __shared__ _Float16 Bs[GT_N * LDT];
#endif
  const int tid  = threadIdx.x;
  const int lane = tid & 31;
  const int wave = tid >> 5;
  const int m0   = blockIdx.x * GT_M;
  const int n0   = blockIdx.y * GT_N;
  const int wrow = (wave & 3) * 32;
  const int wcol = (wave >> 2) * 64;
  const int half = lane >> 4;
  const int ln   = lane & 15;
  const int kb   = half * 8;

  v8f zero8 = {};
  v8f acc[2][4];
  for (int i = 0; i < 2; i++)
    for (int j = 0; j < 4; j++) acc[i][j] = zero8;

  const int lrow = tid >> 1;          // 0..127
  const int lseg = (tid & 1) * 16;    // 0 or 16
  const _Float16* gA = A + (size_t)(m0 + lrow) * HA_D + lseg;
  const _Float16* gB = W + (size_t)(n0 + lrow) * HA_D + lseg;
  const int lofs = lrow * LDT + lseg;

#if HA_ASYNC
  // ---- async DMA double-buffered pipeline: issue(t+1) overlaps compute(t) ----
  ha_async_cp16h(gA, &As[lofs]);
  ha_async_cp16h(gB, &Bs[lofs]);
  for (int it = 0; it < GT_NT; ++it) {
    HA_WAIT_ASYNC();          // this wave's tile-it DMA done (before barrier!)
    __syncthreads();          // all waves' tile-it visible; buf[(it+1)&1] free
    if (it + 1 < GT_NT) {
      const int nb = (it + 1) & 1;
      ha_async_cp16h(gA + (it + 1) * GT_K, &As[nb * GT_M * LDT + lofs]);
      ha_async_cp16h(gB + (it + 1) * GT_K, &Bs[nb * GT_N * LDT + lofs]);
    }
    gemm_step(&As[(it & 1) * GT_M * LDT], &Bs[(it & 1) * GT_N * LDT],
              acc, wrow, wcol, ln, half, kb);
  }
#else
  for (int it = 0; it < GT_NT; ++it) {
    const int kt = it * GT_K;
    {
      const v8h* sa = (const v8h*)(gA + kt);
      v8h a0 = sa[0], a1 = sa[1];
      const v8h* sb = (const v8h*)(gB + kt);
      v8h b0 = sb[0], b1 = sb[1];
      *(v8h*)&As[lofs]     = a0;
      *(v8h*)&As[lofs + 8] = a1;
      *(v8h*)&Bs[lofs]     = b0;
      *(v8h*)&Bs[lofs + 8] = b1;
    }
    if (it + 1 < GT_NT) {
      __builtin_prefetch(gA + kt + GT_K, 0, 1);
      __builtin_prefetch(gB + kt + GT_K, 0, 1);
    }
    __syncthreads();
    gemm_step(As, Bs, acc, wrow, wcol, ln, half, kb);
    __syncthreads();
  }
#endif

  for (int rt = 0; rt < 2; rt++) {
    for (int ct = 0; ct < 4; ct++) {
      const int mg_base = m0 + wrow + rt * 16 + half * 8;
      const int ng      = n0 + wcol + ct * 16 + ln;
      for (int r = 0; r < 8; r++) {
        const int mg = mg_base + r;
        const float v = acc[rt][ct][r];
        if (mode == 2) {
          ((float*)dst)[(size_t)mg * HA_D + ng] = v;
        } else {
          const int b = mg >> 11, seq = mg & (HA_N - 1);
          const int h = ng >> 6,  d   = ng & (HA_DH - 1);
          if (mode == 0)
            ((_Float16*)dst)[(((size_t)b * HA_H + h) * HA_N + seq) * HA_DH + d] = (_Float16)v;
          else
            ((_Float16*)dst)[(((size_t)b * HA_H + h) * HA_DH + d) * HA_N + seq] = (_Float16)v;
        }
      }
    }
  }
}

// ---------------------------------------------------------------- fused flash attention
// grid: (N/64 row-tiles, B*H). block: 128 threads = 4 waves x 16 rows.
#define FT (64 * 72)   // halves per K/V tile buffer

__global__ __launch_bounds__(128)
void hyp_flash(const _Float16* __restrict__ qh, const _Float16* __restrict__ kh,
               const _Float16* __restrict__ vT, const float* __restrict__ qn,
               const float* __restrict__ kn, _Float16* __restrict__ attn,
               const float* __restrict__ log_c, const float* __restrict__ log_beta) {
#if HA_ASYNC
  __shared__ _Float16 Ks[2 * FT];
  __shared__ _Float16 Vs[2 * FT];
#else
  __shared__ _Float16 Ks[FT];
  __shared__ _Float16 Vs[FT];
#endif
  __shared__ _Float16 Ps[4 * 16 * 72];

  const int tid  = threadIdx.x;
  const int lane = tid & 31, wave = tid >> 5;
  const int half = lane >> 4, ln = lane & 15;
  const int kb   = half * 8;
  const int bh   = blockIdx.y;
  const int tileX = blockIdx.x;
  const int nq_base = tileX * 64 + wave * 16;

  const float c    = log1pf(__expf(log_c[0]));
  const float beta = log1pf(__expf(log_beta[0])) + 0.5f;
  const float NEG  = -3.0e38f;

  const _Float16* qrow = qh + ((size_t)bh * HA_N + nq_base + ln) * HA_DH;
  const v16h Aq0 = cat8(*(const v8h*)(qrow + kb),      *(const v8h*)(qrow + kb + 16));
  const v16h Aq1 = cat8(*(const v8h*)(qrow + 32 + kb), *(const v8h*)(qrow + 32 + kb + 16));

  float qn_r[8], m_i[8], l_i[8];
  for (int r = 0; r < 8; r++) {
    qn_r[r] = qn[(size_t)bh * HA_N + nq_base + r + half * 8];
    m_i[r] = -3.0e38f; l_i[r] = 0.f;
  }
  v8f zero8 = {};
  v8f acc[4];
  for (int dt = 0; dt < 4; dt++) acc[dt] = zero8;

#if HA_ASYNC
  // prologue: issue tile 0 into buffer 0
  for (int u = tid; u < 256; u += 128) {
    const int row = u >> 2, seg = (u & 3) * 16;
    ha_async_cp16h(kh + ((size_t)bh * HA_N + row) * HA_DH + seg, &Ks[row * 72 + seg]);
    ha_async_cp16h(vT + ((size_t)bh * HA_DH + row) * HA_N + seg, &Vs[row * 72 + seg]);
  }
#endif

  for (int ct = 0; ct <= tileX; ct++) {
    const int nk0 = ct * 64;
#if HA_ASYNC
    HA_WAIT_ASYNC();          // own tile-ct DMA complete (before barrier)
    __syncthreads();          // tile ct visible to all; other buffer free
    if (ct < tileX) {
      const int nb = (ct + 1) & 1;
      const int nk1 = nk0 + 64;
      for (int u = tid; u < 256; u += 128) {
        const int row = u >> 2, seg = (u & 3) * 16;
        ha_async_cp16h(kh + ((size_t)bh * HA_N + nk1 + row) * HA_DH + seg,
                       &Ks[nb * FT + row * 72 + seg]);
        ha_async_cp16h(vT + ((size_t)bh * HA_DH + row) * HA_N + nk1 + seg,
                       &Vs[nb * FT + row * 72 + seg]);
      }
    }
    const _Float16* Kb = &Ks[(ct & 1) * FT];
    const _Float16* Vb = &Vs[(ct & 1) * FT];
#else
    for (int u = tid; u < 256; u += 128) {
      const int row = u >> 2, seg = (u & 3) * 16;
      const v8h* sk = (const v8h*)(kh + ((size_t)bh * HA_N + nk0 + row) * HA_DH + seg);
      v8h k0 = sk[0], k1 = sk[1];
      const v8h* sv = (const v8h*)(vT + ((size_t)bh * HA_DH + row) * HA_N + nk0 + seg);
      v8h v0 = sv[0], v1 = sv[1];
      *(v8h*)&Ks[row * 72 + seg]     = k0;
      *(v8h*)&Ks[row * 72 + seg + 8] = k1;
      *(v8h*)&Vs[row * 72 + seg]     = v0;
      *(v8h*)&Vs[row * 72 + seg + 8] = v1;
    }
    __syncthreads();
    const _Float16* Kb = Ks;
    const _Float16* Vb = Vs;
#endif

    // S = Q Kt, hyperbolic score, causal mask
    v8f st[4];
    for (int jb = 0; jb < 4; jb++) {
      const _Float16* kp = &Kb[(jb * 16 + ln) * 72 + half * 8];
      v16h Bk0 = cat8(*(const v8h*)kp,        *(const v8h*)(kp + 16));
      v16h Bk1 = cat8(*(const v8h*)(kp + 32), *(const v8h*)(kp + 48));
      v8f s = zero8;
      s = wmma_f16(Aq0, Bk0, s);
      s = wmma_f16(Aq1, Bk1, s);
      const int   nkcol = nk0 + jb * 16 + ln;
      const float knv   = kn[(size_t)bh * HA_N + nkcol];
      for (int r = 0; r < 8; r++) {
        const int nqrow = nq_base + r + half * 8;
        const float sum_n = qn_r[r] + knv;
        const float dsq   = fmaxf(sum_n - 2.0f * s[r], 0.0f);
        const float dist  = sqrtf(dsq + 1e-8f) * sqrtf(1.0f + c * sum_n);
        s[r] = (nkcol <= nqrow) ? (-beta * dist) : NEG;
      }
      st[jb] = s;
    }

    // online softmax (wave32: xor masks 1..8 reduce the 16-lane column groups)
    float scale_r[8];
    for (int r = 0; r < 8; r++) {
      float mx = fmaxf(fmaxf(st[0][r], st[1][r]), fmaxf(st[2][r], st[3][r]));
      for (int off = 1; off < 16; off <<= 1) mx = fmaxf(mx, __shfl_xor(mx, off, 32));
      const float mn = fmaxf(m_i[r], mx);
      scale_r[r] = __expf(m_i[r] - mn);
      m_i[r] = mn;
      float rs = 0.f;
      for (int jb = 0; jb < 4; jb++) {
        const float p = __expf(st[jb][r] - mn);
        st[jb][r] = p;
        rs += p;
      }
      for (int off = 1; off < 16; off <<= 1) rs += __shfl_xor(rs, off, 32);
      l_i[r] = l_i[r] * scale_r[r] + rs;
    }
    for (int dt = 0; dt < 4; dt++)
      for (int r = 0; r < 8; r++) acc[dt][r] *= scale_r[r];

    // stage P through LDS (C layout -> A layout), wave-local, in-order
    _Float16* Pw = &Ps[wave * 16 * 72];
    for (int jb = 0; jb < 4; jb++)
      for (int r = 0; r < 8; r++)
        Pw[(r + half * 8) * 72 + jb * 16 + ln] = (_Float16)st[jb][r];

    {
      const _Float16* pp = &Pw[ln * 72 + kb];
      v16h Ap0 = cat8(*(const v8h*)pp,        *(const v8h*)(pp + 16));
      v16h Ap1 = cat8(*(const v8h*)(pp + 32), *(const v8h*)(pp + 48));
      for (int dt = 0; dt < 4; dt++) {
        const _Float16* vp = &Vb[(dt * 16 + ln) * 72 + half * 8];
        v16h Bv0 = cat8(*(const v8h*)vp,        *(const v8h*)(vp + 16));
        v16h Bv1 = cat8(*(const v8h*)(vp + 32), *(const v8h*)(vp + 48));
        acc[dt] = wmma_f16(Ap0, Bv0, acc[dt]);
        acc[dt] = wmma_f16(Ap1, Bv1, acc[dt]);
      }
    }
#if !HA_ASYNC
    __syncthreads();
#endif
  }

  const int b = bh >> 3, h = bh & 7;
  for (int dt = 0; dt < 4; dt++) {
    for (int r = 0; r < 8; r++) {
      const int nqrow = nq_base + r + half * 8;
      const float o = acc[dt][r] / l_i[r];
      attn[((size_t)b * HA_N + nqrow) * HA_D + h * HA_DH + dt * 16 + ln] = (_Float16)o;
    }
  }
}

// ---------------------------------------------------------------- launch
extern "C" void kernel_launch(void* const* d_in, const int* in_sizes, int n_in,
                              void* d_out, int out_size, void* d_ws, size_t ws_size,
                              hipStream_t stream) {
  const float* x  = (const float*)d_in[0];
  const float* Wq = (const float*)d_in[1];
  const float* Wk = (const float*)d_in[2];
  const float* Wv = (const float*)d_in[3];
  const float* Wo = (const float*)d_in[4];
  const float* lc = (const float*)d_in[5];
  const float* lb = (const float*)d_in[6];

  const size_t XN = (size_t)HA_BN * HA_D;
  const size_t WN = (size_t)HA_D * HA_D;

  _Float16* xh   = (_Float16*)d_ws;
  _Float16* WqH  = xh  + XN;
  _Float16* WkH  = WqH + WN;
  _Float16* WvH  = WkH + WN;
  _Float16* WoH  = WvH + WN;
  _Float16* qhd  = WoH + WN;   // [b,h,n,dh]
  _Float16* khd  = qhd + XN;   // [b,h,n,dh]
  _Float16* vT   = khd + XN;   // [b,h,dh,n]
  _Float16* attn = vT  + XN;   // [b,n,d]
  float*    qn   = (float*)(attn + XN);
  float*    kn   = qn + (size_t)HA_B * HA_H * HA_N;

  hyp_cast<<<2048, 256, 0, stream>>>(x,  xh,  (int)XN);
  hyp_cast<<<512,  256, 0, stream>>>(Wq, WqH, (int)WN);
  hyp_cast<<<512,  256, 0, stream>>>(Wk, WkH, (int)WN);
  hyp_cast<<<512,  256, 0, stream>>>(Wv, WvH, (int)WN);
  hyp_cast<<<512,  256, 0, stream>>>(Wo, WoH, (int)WN);

  dim3 gg(HA_BN / GT_M, HA_D / GT_N);
  hyp_gemm<<<gg, 256, 0, stream>>>(xh, WqH, qhd, 0);
  hyp_gemm<<<gg, 256, 0, stream>>>(xh, WkH, khd, 0);
  hyp_gemm<<<gg, 256, 0, stream>>>(xh, WvH, vT,  1);

  hyp_norms<<<(HA_B * HA_H * HA_N) / 256, 256, 0, stream>>>(qhd, khd, qn, kn);

  hyp_flash<<<dim3(HA_N / 64, HA_B * HA_H), 128, 0, stream>>>(qhd, khd, vT, qn, kn, attn, lc, lb);

  hyp_gemm<<<gg, 256, 0, stream>>>(attn, WoH, d_out, 2);
}
